// TransformerBlock_Bit_88905823027956
// MI455X (gfx1250) — 16-bit and 8-bit matrix operand order repaired and checked by reading, compiled, not run on hardware
//
#include <hip/hip_runtime.h>
#include <hip/hip_bf16.h>
#include <cstdint>
#include <cstddef>


typedef __attribute__((ext_vector_type(8)))  int    v8i;
typedef __attribute__((ext_vector_type(8)))  float  v8f;
typedef __attribute__((ext_vector_type(16))) __bf16 v16bf;

#define T_TOK 4096
#define C_DIM 1024
#define FF_DIM 4096
#define N_HEADS 16
#define D_HEAD 64


__global__ __launch_bounds__(256) void absmean_kernel(const float* __restrict__ w,
                                                      int n, float* __restrict__ out) {
  __shared__ float red[256];
  float s = 0.f;
  for (int i = threadIdx.x; i < n; i += 256) s += fabsf(w[i]);
  red[threadIdx.x] = s;
  __syncthreads();
  for (int k = 128; k > 0; k >>= 1) {
    if (threadIdx.x < (unsigned)k) red[threadIdx.x] += red[threadIdx.x + k];
    __syncthreads();
  }
  if (threadIdx.x == 0) out[0] = red[0] / (float)n;
}

__global__ __launch_bounds__(256) void sign_kernel(const float* __restrict__ w,
                                                   int8_t* __restrict__ o, int n) {
  int i = blockIdx.x * 256 + threadIdx.x;
  if (i < n) o[i] = (w[i] >= 0.f) ? (int8_t)1 : (int8_t)(-1);
}

__global__ __launch_bounds__(256) void quant_rows_kernel(
    const float* __restrict__ in0, const float* __restrict__ in1,
    const float* __restrict__ gvec, int8_t* __restrict__ outq,
    float* __restrict__ scalesT, int C, int T, int mode) {
  __shared__ float buf[4096];
  __shared__ float red[256];
  __shared__ float sgrp[64];
  const int row = blockIdx.x;
  const int tid = threadIdx.x;
  const float* r0 = in0 + (size_t)row * C;

  float ss = 0.f;
  for (int c = tid; c < C; c += 256) {
    float v;
    if (mode == 2) {
      float u = r0[c];
      float w = in1[(size_t)row * C + c];
      v = (u / (1.f + __expf(-u))) * w;
    } else {
      v = r0[c];
    }
    buf[c] = v;
    ss += v * v;
  }
  if (mode == 1) {
    red[tid] = ss;
    __syncthreads();
    for (int k = 128; k > 0; k >>= 1) {
      if (tid < k) red[tid] += red[tid + k];
      __syncthreads();
    }
    float r = rsqrtf(red[0] / (float)C + 1e-6f);
    for (int c = tid; c < C; c += 256) buf[c] = buf[c] * r * gvec[c];
  }
  __syncthreads();

  const int NG = C >> 6;
  if (tid < NG) {
    float mx = 0.f;
    const float* b = buf + tid * 64;
#pragma unroll 8
    for (int j = 0; j < 64; ++j) mx = fmaxf(mx, fabsf(b[j]));
    mx = fmaxf(mx, 1e-5f);
    float s = mx / 127.f;
    sgrp[tid] = s;
    scalesT[(size_t)tid * T + row] = s;
  }
  __syncthreads();

  for (int c = tid; c < C; c += 256) {
    float q = rintf(buf[c] / sgrp[c >> 6]);
    q = fminf(fmaxf(q, -127.f), 127.f);
    outq[(size_t)row * C + c] = (int8_t)q;
  }
}

template <int EPI>
__global__ __launch_bounds__(256) void bit_gemm_kernel(
    const int8_t* __restrict__ A, const float* __restrict__ scalesT,
    const int8_t* __restrict__ Wsign, const float* __restrict__ alpha_p,
    const float* __restrict__ residual, float* __restrict__ outF,
    __bf16* __restrict__ outB, int M, int N, int K, int ldo) {
  __shared__ __align__(16) int8_t Bs[2][256 * 64];

  const int tid = threadIdx.x;
  const int lane = tid & 31;
  const int wave = tid >> 5;
  const int l15 = lane & 15;
  const int lhi = lane >> 4;
  const int m0 = blockIdx.y * 32 + (wave & 1) * 16;
  const int nblk = blockIdx.x * 256;
  const int G = K >> 6;
  const int rowBase = m0 + lhi * 8;

  const int8_t* Arow = A + (size_t)(m0 + l15) * K + lhi * 16;
  const int8_t* Wrow = Wsign + (size_t)(nblk + tid) * K;
  const unsigned ldsRow0 = (unsigned)(size_t)&Bs[0][tid * 64];
  const unsigned ldsRow1 = (unsigned)(size_t)&Bs[1][tid * 64];

  {
    unsigned long long src = (unsigned long long)Wrow;
#pragma unroll
    for (int c = 0; c < 4; ++c) {
      asm volatile("global_load_async_to_lds_b128 %0, %1, off"
                   :: "v"(ldsRow0 + c * 16), "v"(src + c * 16) : "memory");
    }
  }
  asm volatile("s_wait_asynccnt 0x0" ::: "memory");
  __syncthreads();

  float acc[4][8];
#pragma unroll
  for (int t = 0; t < 4; ++t)
#pragma unroll
    for (int i = 0; i < 8; ++i) acc[t][i] = 0.f;

  for (int g = 0; g < G; ++g) {
    if (g + 1 < G) {
      unsigned long long src = (unsigned long long)(Wrow + (g + 1) * 64);
      unsigned dst = ((g + 1) & 1) ? ldsRow1 : ldsRow0;
#pragma unroll
      for (int c = 0; c < 4; ++c) {
        asm volatile("global_load_async_to_lds_b128 %0, %1, off"
                     :: "v"(dst + c * 16), "v"(src + c * 16) : "memory");
      }
    }

    const int8_t* ag = Arow + g * 64;
    int2 p0 = *(const int2*)(ag);
    int2 p1 = *(const int2*)(ag + 8);
    int2 p2 = *(const int2*)(ag + 32);
    int2 p3 = *(const int2*)(ag + 40);
    v8i a;
    a[0] = p0.x; a[1] = p0.y; a[2] = p1.x; a[3] = p1.y;
    a[4] = p2.x; a[5] = p2.y; a[6] = p3.x; a[7] = p3.y;

    float4 s0 = *(const float4*)(scalesT + (size_t)g * M + rowBase);
    float4 s1 = *(const float4*)(scalesT + (size_t)g * M + rowBase + 4);
    float sv[8] = {s0.x, s0.y, s0.z, s0.w, s1.x, s1.y, s1.z, s1.w};

    if (g + 1 < G) __builtin_prefetch(ag + 64, 0, 0);

    const int8_t* Bcur = &Bs[g & 1][0];
#pragma unroll
    for (int nt = 0; nt < 4; ++nt) {
      const int8_t* bg = Bcur + ((wave >> 1) * 64 + nt * 16 + l15) * 64 + lhi * 16;
      int4 b0 = *(const int4*)(bg);
      int4 b1 = *(const int4*)(bg + 32);
      v8i b;
      b[0] = b0.x; b[1] = b0.y; b[2] = b0.z; b[3] = b0.w;
      b[4] = b1.x; b[5] = b1.y; b[6] = b1.z; b[7] = b1.w;
      v8i cz = {0, 0, 0, 0, 0, 0, 0, 0};
      v8i p = __builtin_amdgcn_wmma_i32_16x16x64_iu8(true, a, true, b, cz, false, false);
#pragma unroll
      for (int i = 0; i < 8; ++i) acc[nt][i] += (float)p[i] * sv[i];
    }

    asm volatile("s_wait_asynccnt 0x0" ::: "memory");
    __syncthreads();
  }

  const float alpha = *alpha_p;
#pragma unroll
  for (int nt = 0; nt < 4; ++nt) {
#pragma unroll
    for (int i = 0; i < 8; ++i) {
      const int row = rowBase + i;
      const int col = nblk + (wave >> 1) * 64 + nt * 16 + l15;
      const float val = acc[nt][i] * alpha;
      if (EPI == 0) {
        outF[(size_t)row * ldo + col] = val;
      } else if (EPI == 1) {
        outB[(size_t)row * ldo + col] = (__bf16)val;
      } else if (EPI == 2) {
        outB[(size_t)col * ldo + row] = (__bf16)val;
      } else {
        outF[(size_t)row * ldo + col] = residual[(size_t)row * ldo + col] + val;
      }
    }
  }
}

__global__ __launch_bounds__(128) void attention_kernel(
    const __bf16* __restrict__ Q, const __bf16* __restrict__ Km,
    const __bf16* __restrict__ Vt, float* __restrict__ Y, int T) {
  const int h = blockIdx.y;
  const int qt = blockIdx.x;
  const int lane = threadIdx.x & 31;
  const int wave = threadIdx.x >> 5;
  const int l15 = lane & 15;
  const int lhi = lane >> 4;
  const int q0 = qt * 64 + wave * 16;

  __shared__ __align__(16) __bf16 Pb[4][16 * 64];

  union AF { v16bf v; int i[8]; };
  union BF { v16bf v; int4 q[2]; };

  AF aq[2];
  {
    const __bf16* qrow = Q + (size_t)(q0 + l15) * C_DIM + h * D_HEAD;
#pragma unroll
    for (int c = 0; c < 2; ++c)
#pragma unroll
      for (int j = 0; j < 8; ++j) {
        int kk = j * 2 + lhi * 16;
        aq[c].i[j] = *(const int*)(qrow + c * 32 + kk);
      }
  }

  float m_[8], l_[8];
  v8f o[4];
#pragma unroll
  for (int i = 0; i < 8; ++i) { m_[i] = -3.0e38f; l_[i] = 0.f; }
#pragma unroll
  for (int dt = 0; dt < 4; ++dt)
#pragma unroll
    for (int i = 0; i < 8; ++i) o[dt][i] = 0.f;

  for (int kb = 0; kb <= qt; ++kb) {
    v8f s[4];
#pragma unroll
    for (int nt = 0; nt < 4; ++nt) {
      const __bf16* krow =
          Km + (size_t)(kb * 64 + nt * 16 + l15) * C_DIM + h * D_HEAD + lhi * 16;
      BF kf0, kf1;
      kf0.q[0] = *(const int4*)(krow);
      kf0.q[1] = *(const int4*)(krow + 8);
      kf1.q[0] = *(const int4*)(krow + 32);
      kf1.q[1] = *(const int4*)(krow + 40);
      v8f z = {0.f, 0.f, 0.f, 0.f, 0.f, 0.f, 0.f, 0.f};
      v8f acc = __builtin_amdgcn_wmma_f32_16x16x32_bf16(false, aq[0].v, false, kf0.v,
                                                        (short)0, z, false, false);
      s[nt] = __builtin_amdgcn_wmma_f32_16x16x32_bf16(false, aq[1].v, false, kf1.v,
                                                      (short)0, acc, false, false);
    }

#pragma unroll
    for (int nt = 0; nt < 4; ++nt)
#pragma unroll
      for (int i = 0; i < 8; ++i) {
        float val = s[nt][i] * 0.125f;
        if (kb == qt) {
          int key = kb * 64 + nt * 16 + l15;
          int qr = q0 + lhi * 8 + i;
          if (key > qr) val = -1.0e30f;
        }
        s[nt][i] = val;
      }

#pragma unroll
    for (int i = 0; i < 8; ++i) {
      float rm = fmaxf(fmaxf(s[0][i], s[1][i]), fmaxf(s[2][i], s[3][i]));
      rm = fmaxf(rm, __shfl_xor(rm, 1));
      rm = fmaxf(rm, __shfl_xor(rm, 2));
      rm = fmaxf(rm, __shfl_xor(rm, 4));
      rm = fmaxf(rm, __shfl_xor(rm, 8));
      float mn = fmaxf(m_[i], rm);
      float corr = __expf(m_[i] - mn);
      m_[i] = mn;
      float rs = 0.f;
#pragma unroll
      for (int nt = 0; nt < 4; ++nt) {
        float p = __expf(s[nt][i] - mn);
        s[nt][i] = p;
        rs += p;
      }
      rs += __shfl_xor(rs, 1);
      rs += __shfl_xor(rs, 2);
      rs += __shfl_xor(rs, 4);
      rs += __shfl_xor(rs, 8);
      l_[i] = l_[i] * corr + rs;
#pragma unroll
      for (int dt = 0; dt < 4; ++dt) o[dt][i] *= corr;
    }

#pragma unroll
    for (int nt = 0; nt < 4; ++nt)
#pragma unroll
      for (int i = 0; i < 8; ++i)
        Pb[wave][(lhi * 8 + i) * 64 + nt * 16 + l15] = (__bf16)s[nt][i];
    __syncthreads();

    AF pa[2];
    {
      const __bf16* prow = &Pb[wave][l15 * 64];
#pragma unroll
      for (int c = 0; c < 2; ++c)
#pragma unroll
        for (int j = 0; j < 8; ++j) {
          int kk = j * 2 + lhi * 16;
          pa[c].i[j] = *(const int*)(prow + c * 32 + kk);
        }
    }

#pragma unroll
    for (int dt = 0; dt < 4; ++dt) {
      const __bf16* vrow =
          Vt + (size_t)(h * D_HEAD + dt * 16 + l15) * T + kb * 64 + lhi * 16;
      BF vf0, vf1;
      vf0.q[0] = *(const int4*)(vrow);
      vf0.q[1] = *(const int4*)(vrow + 8);
      vf1.q[0] = *(const int4*)(vrow + 32);
      vf1.q[1] = *(const int4*)(vrow + 40);
      o[dt] = __builtin_amdgcn_wmma_f32_16x16x32_bf16(false, pa[0].v, false, vf0.v,
                                                      (short)0, o[dt], false, false);
      o[dt] = __builtin_amdgcn_wmma_f32_16x16x32_bf16(false, pa[1].v, false, vf1.v,
                                                      (short)0, o[dt], false, false);
    }
    __syncthreads();
  }

#pragma unroll
  for (int i = 0; i < 8; ++i) l_[i] = 1.f / l_[i];
#pragma unroll
  for (int dt = 0; dt < 4; ++dt)
#pragma unroll
    for (int i = 0; i < 8; ++i) {
      int row = q0 + lhi * 8 + i;
      int col = h * D_HEAD + dt * 16 + l15;
      Y[(size_t)row * C_DIM + col] = o[dt][i] * l_[i];
    }
}

extern "C" void kernel_launch(void* const* d_in, const int* in_sizes, int n_in,
                              void* d_out, int out_size, void* d_ws, size_t ws_size,
                              hipStream_t stream) {
  (void)in_sizes; (void)n_in; (void)out_size; (void)ws_size;
  const float* x  = (const float*)d_in[0];
  const float* wf[7] = {(const float*)d_in[1], (const float*)d_in[2],
                        (const float*)d_in[3], (const float*)d_in[4],
                        (const float*)d_in[5], (const float*)d_in[6],
                        (const float*)d_in[7]};
  const float* g1 = (const float*)d_in[8];
  const float* g2 = (const float*)d_in[9];

  const int wN[7] = {C_DIM, C_DIM, C_DIM, C_DIM, FF_DIM, FF_DIM, C_DIM};
  const int wK[7] = {C_DIM, C_DIM, C_DIM, C_DIM, C_DIM, C_DIM, FF_DIM};

  char* p = (char*)d_ws;
  auto take = [&](size_t bytes) {
    char* r = p;
    p += (bytes + 255) & ~(size_t)255;
    return r;
  };
  int8_t* sgn[7];
  for (int i = 0; i < 7; ++i) sgn[i] = (int8_t*)take((size_t)wN[i] * wK[i]);
  float*  alphas = (float*)take(7 * sizeof(float));
  int8_t* hq     = (int8_t*)take((size_t)T_TOK * FF_DIM);
  float*  scT    = (float*)take((size_t)64 * T_TOK * sizeof(float));
  __bf16* qb     = (__bf16*)take((size_t)T_TOK * C_DIM * 2);
  __bf16* kbm    = (__bf16*)take((size_t)T_TOK * C_DIM * 2);
  __bf16* vT     = (__bf16*)take((size_t)C_DIM * T_TOK * 2);
  float*  y      = (float*)take((size_t)T_TOK * C_DIM * sizeof(float));
  float*  x1     = (float*)take((size_t)T_TOK * C_DIM * sizeof(float));
  float*  u      = (float*)take((size_t)T_TOK * FF_DIM * sizeof(float));
  float*  vv     = (float*)take((size_t)T_TOK * FF_DIM * sizeof(float));

  for (int i = 0; i < 7; ++i) {
    int n = wN[i] * wK[i];
    absmean_kernel<<<1, 256, 0, stream>>>(wf[i], n, alphas + i);
    sign_kernel<<<(n + 255) / 256, 256, 0, stream>>>(wf[i], sgn[i], n);
  }

  quant_rows_kernel<<<T_TOK, 256, 0, stream>>>(x, nullptr, g1, hq, scT, C_DIM, T_TOK, 1);
  dim3 gP(C_DIM / 256, T_TOK / 32);
  bit_gemm_kernel<1><<<gP, 256, 0, stream>>>(hq, scT, sgn[0], alphas + 0, nullptr,
                                             nullptr, qb, T_TOK, C_DIM, C_DIM, C_DIM);
  bit_gemm_kernel<1><<<gP, 256, 0, stream>>>(hq, scT, sgn[1], alphas + 1, nullptr,
                                             nullptr, kbm, T_TOK, C_DIM, C_DIM, C_DIM);
  bit_gemm_kernel<2><<<gP, 256, 0, stream>>>(hq, scT, sgn[2], alphas + 2, nullptr,
                                             nullptr, vT, T_TOK, C_DIM, C_DIM, T_TOK);

  attention_kernel<<<dim3(T_TOK / 64, N_HEADS), 128, 0, stream>>>(qb, kbm, vT, y, T_TOK);

  quant_rows_kernel<<<T_TOK, 256, 0, stream>>>(y, nullptr, nullptr, hq, scT, C_DIM, T_TOK, 0);
  bit_gemm_kernel<3><<<gP, 256, 0, stream>>>(hq, scT, sgn[3], alphas + 3, x,
                                             x1, nullptr, T_TOK, C_DIM, C_DIM, C_DIM);

  quant_rows_kernel<<<T_TOK, 256, 0, stream>>>(x1, nullptr, g2, hq, scT, C_DIM, T_TOK, 1);
  dim3 gF(FF_DIM / 256, T_TOK / 32);
  bit_gemm_kernel<0><<<gF, 256, 0, stream>>>(hq, scT, sgn[4], alphas + 4, nullptr,
                                             u, nullptr, T_TOK, FF_DIM, C_DIM, FF_DIM);
  bit_gemm_kernel<0><<<gF, 256, 0, stream>>>(hq, scT, sgn[5], alphas + 5, nullptr,
                                             vv, nullptr, T_TOK, FF_DIM, C_DIM, FF_DIM);
  quant_rows_kernel<<<T_TOK, 256, 0, stream>>>(u, vv, nullptr, hq, scT, FF_DIM, T_TOK, 2);
  bit_gemm_kernel<3><<<gP, 256, 0, stream>>>(hq, scT, sgn[6], alphas + 6, x1,
                                             (float*)d_out, nullptr, T_TOK, C_DIM, FF_DIM, C_DIM);
}
